// PFLM_58815282152097
// MI455X (gfx1250) — matrix operand order repaired and checked by reading, not compiled, not run on hardware
//
#include <hip/hip_runtime.h>
#include <hip/hip_bf16.h>
#include <math.h>

typedef __attribute__((ext_vector_type(16))) __bf16 v16bf;
typedef __attribute__((ext_vector_type(8)))  __bf16 v8bf;
typedef __attribute__((ext_vector_type(8)))  float  v8f;

#define SEQL 64
#define BSZ  32
#define NPART 8
#define BPROWS 256
#define NINP 512
#define NHID 512
#define ZDIM 128
#define NTOK 32000
#define LSE_CHUNKS 50
#define LSE_TILES  40   // 50 chunks * 40 tiles * 16 cols = 32000

// ---------------- device helpers ----------------

__device__ __forceinline__ __bf16 to_bf16(float f) {
  unsigned u = __float_as_uint(f);
  unsigned short h = (unsigned short)((u + 0x7fffu + ((u >> 16) & 1u)) >> 16);
  return __builtin_bit_cast(__bf16, h);
}

__device__ __forceinline__ float sigf(float x) { return 1.0f / (1.0f + expf(-x)); }

__device__ __forceinline__ unsigned hash3(unsigned a, unsigned b, unsigned c) {
  unsigned h = a * 0x9E3779B1u;
  h ^= (b + 0x85EBCA77u) * 0xC2B2AE3Du;
  h ^= (c + 0x27D4EB2Fu) * 0x165667B1u;
  h ^= h >> 15; h *= 0x2C1B3C6Du;
  h ^= h >> 12; h *= 0x297A2D39u;
  h ^= h >> 15;
  return h;
}

// CDNA5 async global->LDS copy of 32 bytes (2x b128), tracked by ASYNCcnt
__device__ __forceinline__ void async_copy32(unsigned ldsOff, const void* gsrc) {
  asm volatile(
      "global_load_async_to_lds_b128 %0, %1, off\n"
      "global_load_async_to_lds_b128 %0, %1, off offset:16"
      :: "v"(ldsOff), "v"(gsrc) : "memory");
}

__device__ __forceinline__ void wait_asynccnt0() {
  asm volatile("s_wait_asynccnt 0x0" ::: "memory");
}

// ---------------- elementwise / utility kernels ----------------

__global__ void k_zero32(unsigned* __restrict__ p, int n) {
  int i = blockIdx.x * blockDim.x + threadIdx.x;
  if (i < n) p[i] = 0u;
}

__global__ void k_setvalf(float* __restrict__ p, float v, int n) {
  int i = blockIdx.x * blockDim.x + threadIdx.x;
  if (i < n) p[i] = v;
}

__global__ void k_cvt_bf16(const float* __restrict__ s, __bf16* __restrict__ d, int n) {
  int i = blockIdx.x * blockDim.x + threadIdx.x;
  if (i < n) d[i] = to_bf16(s[i]);
}

// C[m*N+n] = b0[n] + (b1 ? b1[n] : 0), total = M*N threads
__global__ void k_setbias2(float* __restrict__ C, const float* __restrict__ b0,
                           const float* __restrict__ b1, int N, int total) {
  int i = blockIdx.x * blockDim.x + threadIdx.x;
  if (i >= total) return;
  int n = i % N;
  float v = b0[n];
  if (b1) v += b1[n];
  C[i] = v;
}

// C[r*N+j] = src[(r%32)*N + j], r in [0,256)
__global__ void k_tilerows(float* __restrict__ C, const float* __restrict__ src, int N) {
  int i = blockIdx.x * blockDim.x + threadIdx.x;
  int total = BPROWS * N;
  if (i >= total) return;
  int r = i / N, j = i - r * N;
  C[i] = src[(r & 31) * N + j];
}

// out[row*dim + j] = bf16(table[tok[row]*dim + j])
__global__ void k_tokgather(const int* __restrict__ tok, const float* __restrict__ table,
                            __bf16* __restrict__ out, int dim) {
  int row = blockIdx.x;
  int t = tok[row];
  const float* src = table + (size_t)t * dim;
  __bf16* dst = out + (size_t)row * dim;
  for (int j = threadIdx.x; j < dim; j += blockDim.x) dst[j] = to_bf16(src[j]);
}

// ---------------- WMMA GEMM: C += A(bf16[M,lda]+kA) x B(bf16[N,ldb]+kB)^T ----------------
// grid = (M/16, N/16), block = 32 (one wave per 16x16 tile)

__global__ __launch_bounds__(32) void k_gemm_acc_bf16(
    const __bf16* __restrict__ A, int lda, int kA,
    const __bf16* __restrict__ B, int ldb, int kB,
    float* __restrict__ C, int ldc, int K) {
  int lane = threadIdx.x;
  int lh = lane & 15, sel = lane >> 4;
  int m0 = blockIdx.x << 4, n0 = blockIdx.y << 4;
  const __bf16* arow = A + (size_t)(m0 + lh) * lda + kA;
  const __bf16* brow = B + (size_t)(n0 + lh) * ldb + kB;
  float* cp = C + (size_t)(m0 + sel * 8) * ldc + n0 + lh;
  v8f acc;
#pragma unroll
  for (int r = 0; r < 8; ++r) acc[r] = cp[r * ldc];
  for (int k = 0; k < K; k += 32) {
    v16bf av = *(const v16bf*)(arow + k + sel * 16);
    v16bf bv = *(const v16bf*)(brow + k + sel * 16);
    acc = __builtin_amdgcn_wmma_f32_16x16x32_bf16(false, av, false, bv, (short)0, acc,
                                                  false, false);
  }
#pragma unroll
  for (int r = 0; r < 8; ++r) cp[r * ldc] = acc[r];
}

// ---------------- fused logits + online logsumexp ----------------
// A = dh_bf16 [256,512]; B = Wout_bf16 [32000,512].
// One 512-thread block (16 waves) covers all 16 row-tiles; each Wout tile is
// async-staged into LDS once (double-buffered) and shared by all waves.
// grid = (LSE_CHUNKS) blocks.

__global__ __launch_bounds__(512) void k_logits_lse(
    const __bf16* __restrict__ A, const __bf16* __restrict__ B,
    const float* __restrict__ bout, float* __restrict__ pmax, float* __restrict__ psum) {
  __shared__ __bf16 sB[2 * 16 * NHID];  // 2 x 16KB double buffer
  int tid = threadIdx.x;
  int wave = tid >> 5, lane = tid & 31;
  int lh = lane & 15, sel = lane >> 4;
  int chunk = blockIdx.x;
  int m0 = wave << 4;

  // preload this wave's A tile (16 rows x K=512) into registers
  const __bf16* arow = A + (size_t)(m0 + lh) * NHID;
  v16bf areg[16];
#pragma unroll
  for (int c = 0; c < 16; ++c) {
    areg[c] = *(const v16bf*)(arow + c * 32 + sel * 16);
  }

  float rmax[8], rsum[8];
#pragma unroll
  for (int r = 0; r < 8; ++r) { rmax[r] = -3.4e38f; rsum[r] = 0.0f; }

  // stage tile 0: 16 rows of Wout are contiguous (16KB); 512 thr x 32B each
  {
    int n0 = (chunk * LSE_TILES) << 4;
    const __bf16* gsrc = B + (size_t)n0 * NHID + tid * 16;
    unsigned dst = (unsigned)(size_t)(&sB[0]) + (unsigned)(tid * 32);
    async_copy32(dst, gsrc);
  }
  wait_asynccnt0();
  __syncthreads();

  int buf = 0;
  for (int j = 0; j < LSE_TILES; ++j) {
    if (j + 1 < LSE_TILES) {
      int n1 = (chunk * LSE_TILES + j + 1) << 4;
      const __bf16* gsrc = B + (size_t)n1 * NHID + tid * 16;
      unsigned dst = (unsigned)(size_t)(&sB[(buf ^ 1) * 16 * NHID]) + (unsigned)(tid * 32);
      async_copy32(dst, gsrc);
    }
    const __bf16* bbase = &sB[buf * 16 * NHID + lh * NHID + sel * 16];
    v8f acc;
#pragma unroll
    for (int r = 0; r < 8; ++r) acc[r] = 0.0f;
#pragma unroll
    for (int c = 0; c < 16; ++c) {
      v16bf bv = *(const v16bf*)(bbase + c * 32);
      acc = __builtin_amdgcn_wmma_f32_16x16x32_bf16(false, areg[c], false, bv, (short)0,
                                                    acc, false, false);
    }
    int n0 = (chunk * LSE_TILES + j) << 4;
    float bb = bout[n0 + lh];
#pragma unroll
    for (int r = 0; r < 8; ++r) {
      float v = acc[r] + bb;
      if (v > rmax[r]) {
        rsum[r] = rsum[r] * expf(rmax[r] - v) + 1.0f;
        rmax[r] = v;
      } else {
        rsum[r] += expf(v - rmax[r]);
      }
    }
    wait_asynccnt0();
    __syncthreads();
    buf ^= 1;
  }

  // merge across the 16 lanes of each half-wave
#pragma unroll
  for (int r = 0; r < 8; ++r) {
    float m = rmax[r], sv = rsum[r];
#pragma unroll
    for (int off = 1; off < 16; off <<= 1) {
      float om = __shfl_xor(m, off, 32);
      float os = __shfl_xor(sv, off, 32);
      float nm = fmaxf(m, om);
      sv = sv * expf(m - nm) + os * expf(om - nm);
      m = nm;
    }
    rmax[r] = m; rsum[r] = sv;
  }
  if (lh == 0) {
    int rb = m0 + sel * 8;
#pragma unroll
    for (int r = 0; r < 8; ++r) {
      pmax[(rb + r) * LSE_CHUNKS + chunk] = rmax[r];
      psum[(rb + r) * LSE_CHUNKS + chunk] = rsum[r];
    }
  }
}

// logit at target token (f32 precision), one block per row
__global__ void k_tgt_logit(const float* __restrict__ dh, const float* __restrict__ wout,
                            const float* __restrict__ bout, const int* __restrict__ tok,
                            float* __restrict__ out) {
  __shared__ float sred[128];
  int r = blockIdx.x, tid = threadIdx.x;
  int tg = tok[r & 31];
  const float* a = dh + (size_t)r * NHID;
  const float* w = wout + (size_t)tg * NHID;
  float s = 0.0f;
  for (int j = tid; j < NHID; j += 128) s += a[j] * w[j];
  sred[tid] = s;
  __syncthreads();
  for (int st = 64; st > 0; st >>= 1) {
    if (tid < st) sred[tid] += sred[tid + st];
    __syncthreads();
  }
  if (tid == 0) out[r] = sred[0] + bout[tg];
}

// combine LSE partials, NLL[r] = lse - tlogit[r]; nll_acc += sum(NLL)/P  (one block, 256 thr)
__global__ void k_fin_nll(const float* __restrict__ pmax, const float* __restrict__ psum,
                          const float* __restrict__ tlog, float* __restrict__ NLL,
                          float* __restrict__ d_nll) {
  __shared__ float sred[256];
  int r = threadIdx.x;
  float m = -3.4e38f, s = 0.0f;
  for (int c = 0; c < LSE_CHUNKS; ++c) {
    float cm = pmax[r * LSE_CHUNKS + c];
    float cs = psum[r * LSE_CHUNKS + c];
    float nm = fmaxf(m, cm);
    s = s * expf(m - nm) + cs * expf(cm - nm);
    m = nm;
  }
  float nl = (m + logf(s)) - tlog[r];
  NLL[r] = nl;
  sred[r] = nl;
  __syncthreads();
  for (int st = 128; st > 0; st >>= 1) {
    if (r < st) sred[r] += sred[r + st];
    __syncthreads();
  }
  if (r == 0) *d_nll += sred[0] * (1.0f / NPART);
}

// alpha[r] = -NLL[r] - 0.5*(||z-ph||^2 - ||eps||^2)   (one wave per row)
__global__ __launch_bounds__(32) void k_alpha(const float* __restrict__ z,
                                              const float* __restrict__ ph,
                                              const float* __restrict__ eps_t,
                                              const float* __restrict__ NLL,
                                              float* __restrict__ alpha) {
  int r = blockIdx.x, lane = threadIdx.x;
  float s1 = 0.0f, s2 = 0.0f;
#pragma unroll
  for (int q = 0; q < 4; ++q) {
    int j = lane + q * 32;
    float d = z[r * ZDIM + j] - ph[r * ZDIM + j];
    s1 += d * d;
    float e = eps_t[r * ZDIM + j];
    s2 += e * e;
  }
#pragma unroll
  for (int off = 16; off > 0; off >>= 1) {
    s1 += __shfl_xor(s1, off, 32);
    s2 += __shfl_xor(s2, off, 32);
  }
  if (lane == 0) alpha[r] = -NLL[r] - 0.5f * (s1 - s2);
}

// weight update / ESS / resample-index (one block of 256 = P*B)
__global__ void k_weights(float* __restrict__ accw, const float* __restrict__ alpha,
                          float* __restrict__ d_loss, float* __restrict__ d_res,
                          int* __restrict__ idx, int step) {
  __shared__ float swa[256], sp[256], sZ[32], scol[32], sflagv[32];
  __shared__ int sflag;
  int tid = threadIdx.x, p = tid >> 5, b = tid & 31;
  float wa = accw[tid] + alpha[tid];
  swa[tid] = wa;
  __syncthreads();
  if (tid < 32) {
    float m = -3.4e38f;
    for (int q = 0; q < NPART; ++q) m = fmaxf(m, swa[q * 32 + tid]);
    float s = 0.0f;
    for (int q = 0; q < NPART; ++q) s += expf(swa[q * 32 + tid] - m);
    sZ[tid] = m + logf(s);
  }
  __syncthreads();
  float accw2 = wa - sZ[b];
  float prob = expf(accw2) + 0.01f;
  sp[tid] = prob;
  __syncthreads();
  if (tid < 32) {
    float cs = 0.0f;
    for (int q = 0; q < NPART; ++q) cs += sp[q * 32 + tid];
    scol[tid] = cs;
  }
  __syncthreads();
  float pn = sp[tid] / scol[b];
  __syncthreads();
  sp[tid] = pn;
  __syncthreads();
  if (tid < 32) {
    float s2 = 0.0f;
    for (int q = 0; q < NPART; ++q) { float v = sp[q * 32 + tid]; s2 += v * v; }
    float ess = 1.0f / s2;
    sflagv[tid] = (ess * (1.0f / NPART) < 0.6f) ? 1.0f : 0.0f;
  }
  __syncthreads();
  if (tid == 0) {
    int f = 0;
    float ls = 0.0f;
    for (int q = 0; q < 32; ++q) { if (sflagv[q] > 0.0f) f = 1; ls += sZ[q]; }
    sflag = f;
    *d_loss += ls;
    *d_res += (float)f;
  }
  __syncthreads();
  int do_rs = sflag;
  unsigned h = hash3(0xC0FFEEu + (unsigned)step, (unsigned)p, (unsigned)b);
  float u = (float)(h & 0xFFFFFFu) * (1.0f / 16777216.0f);
  float csum = 0.0f;
  int anc = NPART - 1;
  for (int q = 0; q < NPART; ++q) {
    csum += sp[q * 32 + b];
    if (u < csum) { anc = q; break; }
  }
  idx[tid] = do_rs ? (anc * 32 + b) : tid;
  accw[tid] = do_rs ? -2.0794415416798357f /* -ln 8 */ : accw2;
}

// z = mean + eps; also bf16 copy
__global__ void k_zmake(const float* __restrict__ mg, const float* __restrict__ eps_t,
                        float* __restrict__ z, __bf16* __restrict__ zb) {
  int i = blockIdx.x * blockDim.x + threadIdx.x;
  if (i < BPROWS * ZDIM) {
    float v = mg[i] + eps_t[i];
    z[i] = v;
    zb[i] = to_bf16(v);
  }
}

// state permutation gather (always runs; idx==identity when no resample)
__global__ void k_gather(const float* __restrict__ src, float* __restrict__ dst,
                         __bf16* __restrict__ dstb, const int* __restrict__ idx, int dim) {
  int r = blockIdx.x;
  int j = blockIdx.y * 128 + threadIdx.x;
  if (j < dim) {
    float v = src[(size_t)idx[r] * dim + j];
    dst[(size_t)r * dim + j] = v;
    if (dstb) dstb[(size_t)r * dim + j] = to_bf16(v);
  }
}

// fused LSTM cell elementwise: gates [rows,4H] (i,f,g,o), c in-place
__global__ void k_cell(const float* __restrict__ gates, float* __restrict__ c_state,
                       float* __restrict__ h_out, __bf16* __restrict__ h_bf, int H,
                       int rows) {
  int i = blockIdx.x * blockDim.x + threadIdx.x;
  if (i >= rows * H) return;
  int r = i / H, j = i - r * H;
  size_t g0 = (size_t)r * 4 * H + j;
  float gi = gates[g0];
  float gf = gates[g0 + H];
  float gg = gates[g0 + 2 * H];
  float go = gates[g0 + 3 * H];
  float c2 = sigf(gf) * c_state[i] + sigf(gi) * tanhf(gg);
  float h = sigf(go) * tanhf(c2);
  c_state[i] = c2;
  if (h_out) h_out[i] = h;
  if (h_bf) h_bf[i] = to_bf16(h);
}

// Box-Muller normals (deterministic hash RNG)
__global__ void k_epsfill(float* __restrict__ eps, int npairs) {
  int i = blockIdx.x * blockDim.x + threadIdx.x;
  if (i >= npairs) return;
  unsigned h1 = hash3(0x9E37u, (unsigned)i, 1u);
  unsigned h2 = hash3(0x85EBu, (unsigned)i, 2u);
  float u1 = ((float)(h1 >> 8) + 0.5f) * (1.0f / 16777216.0f);
  float u2 = (float)(h2 >> 8) * (1.0f / 16777216.0f);
  float rr = sqrtf(-2.0f * logf(u1));
  float th = 6.28318530718f * u2;
  eps[2 * i] = rr * cosf(th);
  eps[2 * i + 1] = rr * sinf(th);
}

__global__ void k_finout(const float* loss, const float* nll, const float* res,
                         float* __restrict__ out) {
  out[0] = -(*loss);
  out[1] = *nll;
  out[2] = *res;
}

// ---------------- host-side launch helpers ----------------

static inline void gemm(hipStream_t s, const __bf16* A, int lda, int kA, const __bf16* B,
                        int ldb, int kB, float* C, int ldc, int M, int N, int K) {
  dim3 g(M / 16, N / 16);
  k_gemm_acc_bf16<<<g, 32, 0, s>>>(A, lda, kA, B, ldb, kB, C, ldc, K);
}

static inline void cell(hipStream_t s, const float* g, float* c, float* h, __bf16* hb,
                        int H, int rows) {
  int n = rows * H;
  k_cell<<<(n + 127) / 128, 128, 0, s>>>(g, c, h, hb, H, rows);
}

static inline void cvt(hipStream_t s, const float* src, __bf16* dst, int n) {
  k_cvt_bf16<<<(n + 255) / 256, 256, 0, s>>>(src, dst, n);
}

extern "C" void kernel_launch(void* const* d_in, const int* in_sizes, int n_in,
                              void* d_out, int out_size, void* d_ws, size_t ws_size,
                              hipStream_t stream) {
  (void)in_sizes; (void)n_in; (void)out_size; (void)ws_size;
  const int*   tokens  = (const int*)d_in[0];
  const int*   targets = (const int*)d_in[1];
  const float* inp_emb = (const float*)d_in[2];
  const float* dec_emb = (const float*)d_in[3];
  const float* Wih_f = (const float*)d_in[4];
  const float* Whh_f = (const float*)d_in[5];
  const float* bih_f = (const float*)d_in[6];
  const float* bhh_f = (const float*)d_in[7];
  const float* Wih_b = (const float*)d_in[8];
  const float* Whh_b = (const float*)d_in[9];
  const float* bih_b = (const float*)d_in[10];
  const float* bhh_b = (const float*)d_in[11];
  const float* Wz_ih = (const float*)d_in[12];
  const float* Wz_hh = (const float*)d_in[13];
  const float* bz_ih = (const float*)d_in[14];
  const float* bz_hh = (const float*)d_in[15];
  const float* Wp_ih = (const float*)d_in[16];
  const float* Wp_hh = (const float*)d_in[17];
  const float* bp_ih = (const float*)d_in[18];
  const float* bp_hh = (const float*)d_in[19];
  const float* Wd_ih = (const float*)d_in[20];
  const float* Wd_hh = (const float*)d_in[21];
  const float* bd_ih = (const float*)d_in[22];
  const float* bd_hh = (const float*)d_in[23];
  const float* Wm    = (const float*)d_in[24];
  const float* bm    = (const float*)d_in[25];
  const float* Wout  = (const float*)d_in[26];
  const float* bout  = (const float*)d_in[27];

  char* wsb = (char*)d_ws;
  size_t off = 0;
  auto alloc = [&](size_t bytes) -> void* {
    void* p = wsb + off;
    off += (bytes + 255) & ~(size_t)255;
    return p;
  };

  // bf16 weights
  __bf16* wihf = (__bf16*)alloc((size_t)2048 * 512 * 2);
  __bf16* whhf = (__bf16*)alloc((size_t)2048 * 512 * 2);
  __bf16* wihb = (__bf16*)alloc((size_t)2048 * 512 * 2);
  __bf16* whhb = (__bf16*)alloc((size_t)2048 * 512 * 2);
  __bf16* wzih = (__bf16*)alloc((size_t)512 * 1024 * 2);
  __bf16* wzhh = (__bf16*)alloc((size_t)512 * 128 * 2);
  __bf16* wpih = (__bf16*)alloc((size_t)512 * 128 * 2);
  __bf16* wphh = (__bf16*)alloc((size_t)512 * 128 * 2);
  __bf16* wdih = (__bf16*)alloc((size_t)2048 * 640 * 2);
  __bf16* wdhh = (__bf16*)alloc((size_t)2048 * 512 * 2);
  __bf16* wmb  = (__bf16*)alloc((size_t)128 * 128 * 2);
  __bf16* woutb = (__bf16*)alloc((size_t)NTOK * 512 * 2);
  // bf16 activations
  __bf16* embb = (__bf16*)alloc((size_t)2048 * 512 * 2);
  __bf16* oeb  = (__bf16*)alloc((size_t)2048 * 512 * 2);
  __bf16* hsfb = (__bf16*)alloc((size_t)SEQL * 32 * 512 * 2);
  __bf16* hsbb = (__bf16*)alloc((size_t)SEQL * 32 * 512 * 2);
  // f32 precomputed gate contributions
  float* Xf  = (float*)alloc((size_t)2048 * 2048 * 4);
  float* Xb  = (float*)alloc((size_t)2048 * 2048 * 4);
  float* HG  = (float*)alloc((size_t)2048 * 512 * 4);
  float* OEG = (float*)alloc((size_t)2048 * 2048 * 4);
  float* eps = (float*)alloc((size_t)SEQL * BPROWS * ZDIM * 4);

  // ---- zeroed state zone ----
  size_t zone0 = off;
  float* hS[2];  hS[0] = (float*)alloc(BPROWS * ZDIM * 4);  hS[1] = (float*)alloc(BPROWS * ZDIM * 4);
  float* cS[2];  cS[0] = (float*)alloc(BPROWS * ZDIM * 4);  cS[1] = (float*)alloc(BPROWS * ZDIM * 4);
  float* phS[2]; phS[0] = (float*)alloc(BPROWS * ZDIM * 4); phS[1] = (float*)alloc(BPROWS * ZDIM * 4);
  float* pcS[2]; pcS[0] = (float*)alloc(BPROWS * ZDIM * 4); pcS[1] = (float*)alloc(BPROWS * ZDIM * 4);
  float* dhS[2]; dhS[0] = (float*)alloc(BPROWS * NHID * 4); dhS[1] = (float*)alloc(BPROWS * NHID * 4);
  float* dcS[2]; dcS[0] = (float*)alloc(BPROWS * NHID * 4); dcS[1] = (float*)alloc(BPROWS * NHID * 4);
  __bf16* hB[2];  hB[0] = (__bf16*)alloc(BPROWS * ZDIM * 2);  hB[1] = (__bf16*)alloc(BPROWS * ZDIM * 2);
  __bf16* phB[2]; phB[0] = (__bf16*)alloc(BPROWS * ZDIM * 2); phB[1] = (__bf16*)alloc(BPROWS * ZDIM * 2);
  __bf16* dhB[2]; dhB[0] = (__bf16*)alloc(BPROWS * NHID * 2); dhB[1] = (__bf16*)alloc(BPROWS * NHID * 2);
  __bf16* zB = (__bf16*)alloc(BPROWS * ZDIM * 2);
  float* enc_cf = (float*)alloc(32 * NHID * 4);
  float* enc_cb = (float*)alloc(32 * NHID * 4);
  float* d_loss = (float*)alloc(4);
  float* d_nll  = (float*)alloc(4);
  float* d_res  = (float*)alloc(4);
  size_t zone1 = off;

  // scratch (overwritten every step before use)
  float* zf32 = (float*)alloc(BPROWS * ZDIM * 4);
  float* zg = (float*)alloc(BPROWS * 512 * 4);
  float* dg = (float*)alloc((size_t)BPROWS * 2048 * 4);
  float* pg = (float*)alloc(BPROWS * 512 * 4);
  float* mg = (float*)alloc(BPROWS * ZDIM * 4);
  float* pmax = (float*)alloc(BPROWS * LSE_CHUNKS * 4);
  float* psum = (float*)alloc(BPROWS * LSE_CHUNKS * 4);
  float* tlog = (float*)alloc(BPROWS * 4);
  float* NLL  = (float*)alloc(BPROWS * 4);
  float* alph = (float*)alloc(BPROWS * 4);
  float* accw = (float*)alloc(BPROWS * 4);
  int*   ridx = (int*)alloc(BPROWS * 4);

  // ---- init ----
  {
    int nzero = (int)((zone1 - zone0) / 4);
    k_zero32<<<(nzero + 255) / 256, 256, 0, stream>>>((unsigned*)(wsb + zone0), nzero);
    k_setvalf<<<1, 256, 0, stream>>>(accw, -2.0794415416798357f, BPROWS);
  }
  // weights -> bf16
  cvt(stream, Wih_f, wihf, 2048 * 512);  cvt(stream, Whh_f, whhf, 2048 * 512);
  cvt(stream, Wih_b, wihb, 2048 * 512);  cvt(stream, Whh_b, whhb, 2048 * 512);
  cvt(stream, Wz_ih, wzih, 512 * 1024);  cvt(stream, Wz_hh, wzhh, 512 * 128);
  cvt(stream, Wp_ih, wpih, 512 * 128);   cvt(stream, Wp_hh, wphh, 512 * 128);
  cvt(stream, Wd_ih, wdih, 2048 * 640);  cvt(stream, Wd_hh, wdhh, 2048 * 512);
  cvt(stream, Wm, wmb, 128 * 128);       cvt(stream, Wout, woutb, NTOK * 512);
  // embedding gathers
  k_tokgather<<<2048, 128, 0, stream>>>(tokens, inp_emb, embb, NINP);
  k_tokgather<<<2048, 128, 0, stream>>>(targets, dec_emb, oeb, NINP);
  // eps
  k_epsfill<<<(SEQL * BPROWS * ZDIM / 2 + 255) / 256, 256, 0, stream>>>(
      eps, SEQL * BPROWS * ZDIM / 2);

  // ---- encoder input GEMMs: X = bias + emb @ Wih^T ----
  k_setbias2<<<(2048 * 2048 + 255) / 256, 256, 0, stream>>>(Xf, bih_f, bhh_f, 2048, 2048 * 2048);
  gemm(stream, embb, 512, 0, wihf, 512, 0, Xf, 2048, 2048, 2048, 512);
  k_setbias2<<<(2048 * 2048 + 255) / 256, 256, 0, stream>>>(Xb, bih_b, bhh_b, 2048, 2048 * 2048);
  gemm(stream, embb, 512, 0, wihb, 512, 0, Xb, 2048, 2048, 2048, 512);

  // ---- encoder recurrences ----
  for (int t = 0; t < SEQL; ++t) {
    float* g = Xf + (size_t)t * 32 * 2048;
    if (t > 0)
      gemm(stream, hsfb + (size_t)(t - 1) * 32 * 512, 512, 0, whhf, 512, 0, g, 2048, 32,
           2048, 512);
    cell(stream, g, enc_cf, nullptr, hsfb + (size_t)t * 32 * 512, NHID, 32);
  }
  for (int s = 0; s < SEQL; ++s) {
    int t = SEQL - 1 - s;
    float* g = Xb + (size_t)t * 32 * 2048;
    if (s > 0)
      gemm(stream, hsbb + (size_t)(t + 1) * 32 * 512, 512, 0, whhb, 512, 0, g, 2048, 32,
           2048, 512);
    cell(stream, g, enc_cb, nullptr, hsbb + (size_t)t * 32 * 512, NHID, 32);
  }

  // ---- HG = bias_z + [hs_f, hs_b] @ Wz_ih^T ----
  k_setbias2<<<(2048 * 512 + 255) / 256, 256, 0, stream>>>(HG, bz_ih, bz_hh, 512, 2048 * 512);
  gemm(stream, hsfb, 512, 0, wzih, 1024, 0, HG, 512, 2048, 512, 512);
  gemm(stream, hsbb, 512, 0, wzih, 1024, 512, HG, 512, 2048, 512, 512);
  // ---- OEG = bias_d + oe @ Wd_ih[:,128:]^T ----
  k_setbias2<<<(2048 * 2048 + 255) / 256, 256, 0, stream>>>(OEG, bd_ih, bd_hh, 2048, 2048 * 2048);
  gemm(stream, oeb, 512, 0, wdih, 640, 128, OEG, 2048, 2048, 2048, 512);

  // ---- main particle-filter loop ----
  int cur = 0;
  for (int t = 0; t < SEQL; ++t) {
    int nxt = cur ^ 1;
    const float* eps_t = eps + (size_t)t * BPROWS * ZDIM;

    // z-decoder cell
    k_tilerows<<<(BPROWS * 512 + 255) / 256, 256, 0, stream>>>(zg, HG + (size_t)t * 32 * 512, 512);
    gemm(stream, hB[cur], ZDIM, 0, wzhh, 128, 0, zg, 512, BPROWS, 512, 128);
    cell(stream, zg, cS[cur], hS[cur], hB[cur], ZDIM, BPROWS);

    // mean + z
    k_setbias2<<<(BPROWS * ZDIM + 255) / 256, 256, 0, stream>>>(mg, bm, nullptr, ZDIM,
                                                                BPROWS * ZDIM);
    gemm(stream, hB[cur], ZDIM, 0, wmb, 128, 0, mg, ZDIM, BPROWS, ZDIM, ZDIM);
    k_zmake<<<(BPROWS * ZDIM + 255) / 256, 256, 0, stream>>>(mg, eps_t, zf32, zB);

    // decoder cell
    k_tilerows<<<(BPROWS * 2048 + 255) / 256, 256, 0, stream>>>(dg, OEG + (size_t)t * 32 * 2048,
                                                                2048);
    gemm(stream, zB, ZDIM, 0, wdih, 640, 0, dg, 2048, BPROWS, 2048, 128);
    gemm(stream, dhB[cur], NHID, 0, wdhh, 512, 0, dg, 2048, BPROWS, 2048, 512);
    cell(stream, dg, dcS[cur], dhS[cur], dhB[cur], NHID, BPROWS);

    // logits -> logsumexp -> NLL (16 waves/block share async-staged Wout tiles)
    k_logits_lse<<<dim3(LSE_CHUNKS), 512, 0, stream>>>(dhB[cur], woutb, bout, pmax, psum);
    k_tgt_logit<<<BPROWS, 128, 0, stream>>>(dhS[cur], Wout, bout, tokens + t * 32, tlog);
    k_fin_nll<<<1, 256, 0, stream>>>(pmax, psum, tlog, NLL, d_nll);

    // alpha, weights, resample indices
    k_alpha<<<BPROWS, 32, 0, stream>>>(zf32, phS[cur], eps_t, NLL, alph);
    k_weights<<<1, 256, 0, stream>>>(accw, alph, d_loss, d_res, ridx, t);

    // permute states into next buffers
    {
      dim3 gz(BPROWS, 1), gh(BPROWS, 4);
      k_gather<<<gz, 128, 0, stream>>>(hS[cur], hS[nxt], hB[nxt], ridx, ZDIM);
      k_gather<<<gz, 128, 0, stream>>>(cS[cur], cS[nxt], nullptr, ridx, ZDIM);
      k_gather<<<gz, 128, 0, stream>>>(phS[cur], phS[nxt], phB[nxt], ridx, ZDIM);
      k_gather<<<gz, 128, 0, stream>>>(pcS[cur], pcS[nxt], nullptr, ridx, ZDIM);
      k_gather<<<gh, 128, 0, stream>>>(dhS[cur], dhS[nxt], dhB[nxt], ridx, NHID);
      k_gather<<<gh, 128, 0, stream>>>(dcS[cur], dcS[nxt], nullptr, ridx, NHID);
    }

    // AR prior cell: (ph,pc) = lstm(h, ph, pc)
    k_setbias2<<<(BPROWS * 512 + 255) / 256, 256, 0, stream>>>(pg, bp_ih, bp_hh, 512,
                                                               BPROWS * 512);
    gemm(stream, hB[nxt], ZDIM, 0, wpih, 128, 0, pg, 512, BPROWS, 512, 128);
    gemm(stream, phB[nxt], ZDIM, 0, wphh, 128, 0, pg, 512, BPROWS, 512, 128);
    cell(stream, pg, pcS[nxt], phS[nxt], nullptr, ZDIM, BPROWS);

    cur = nxt;
  }

  k_finout<<<1, 1, 0, stream>>>(d_loss, d_nll, d_res, (float*)d_out);
}
